// PointerStandard_9242769621208
// MI455X (gfx1250) — matrix operand order repaired and checked by reading, not compiled, not run on hardware
//
#include <hip/hip_runtime.h>
#include <hip/hip_bf16.h>

// Problem sizes (fixed by the reference)
#define BB    32
#define NSEQ  2048
#define DD    1024
#define ROWS  (BB * NSEQ)          // 65536 GEMM rows
#define ROWSTRIDE (DD + 8)         // padded bf16 row in LDS (2064 B, 16B-aligned, rotates banks)

typedef __attribute__((ext_vector_type(16))) __bf16 v16bf;
typedef __attribute__((ext_vector_type(8)))  __bf16 v8bf;
typedef __attribute__((ext_vector_type(8)))  float  v8f;

#define LOG_EPS (-103.6163292f)    // logf(1e-45f)

__device__ __forceinline__ __bf16 f2bf(float f) { return (__bf16)f; }

// Branch-free tanh: prefer the CDNA5 hardware transcendental (v_tanh_f32).
__device__ __forceinline__ float fast_tanh(float x) {
#if __has_builtin(__builtin_amdgcn_tanhf)
    return __builtin_amdgcn_tanhf(x);
#elif __has_builtin(__builtin_amdgcn_tanh_f32)
    return __builtin_amdgcn_tanh_f32(x);
#elif __has_builtin(__builtin_amdgcn_exp2f) && __has_builtin(__builtin_amdgcn_rcpf)
    // tanh(|x|) = (1-e)/(1+e), e = 2^(-2|x|*log2(e));  ~6 straight-line VALU/TRANS ops
    const float ax = __builtin_fabsf(x);
    const float e  = __builtin_amdgcn_exp2f(ax * -2.885390082f);   // -2*log2(e)
    const float t  = (1.0f - e) * __builtin_amdgcn_rcpf(1.0f + e);
    return __builtin_copysignf(t, x);
#else
    return tanhf(x);
#endif
}

// Assemble a 16-bit A fragment (ISA 16x32 layout): this lane's 16 bf16 values are
// two 8-element (16B) chunks at +0 and +16 elements from its per-lane base.
__device__ __forceinline__ v16bf load_afrag(const __bf16* p) {
    v8bf lo = *(const v8bf*)(p);
    v8bf hi = *(const v8bf*)(p + 16);
    v16bf r;
#pragma unroll
    for (int i = 0; i < 8; ++i) { r[i] = lo[i]; r[i + 8] = hi[i]; }
    return r;
}

// ---------------------------------------------------------------------------
// Kernel 1: W1 (f32, [k][n]) -> W1T (bf16, [n][k]) tiled transpose via LDS.
// ---------------------------------------------------------------------------
__global__ __launch_bounds__(256) void w1_to_bf16_T(const float* __restrict__ W1,
                                                    __bf16* __restrict__ W1T) {
    __shared__ float tile[32][33];
    const int n0 = blockIdx.x * 32;
    const int k0 = blockIdx.y * 32;
#pragma unroll
    for (int j = 0; j < 32; j += 8)
        tile[threadIdx.y + j][threadIdx.x] =
            W1[(size_t)(k0 + threadIdx.y + j) * DD + n0 + threadIdx.x];
    __syncthreads();
#pragma unroll
    for (int j = 0; j < 32; j += 8)
        W1T[(size_t)(n0 + threadIdx.y + j) * DD + k0 + threadIdx.x] =
            f2bf(tile[threadIdx.x][threadIdx.y + j]);
}

// ---------------------------------------------------------------------------
// Kernel 2: dec = decoder_state @ W2   [32,1024] x [1024,1024] (tiny)
// ---------------------------------------------------------------------------
__global__ __launch_bounds__(256) void dec_gemm(const float* __restrict__ ds,
                                                const float* __restrict__ W2,
                                                float* __restrict__ dec) {
    const int b = blockIdx.x;
    const int e = blockIdx.y * 256 + threadIdx.x;
    const float* dsb = ds + (size_t)b * DD;
    float acc = 0.f;
    for (int d = 0; d < DD; ++d)
        acc = fmaf(dsb[d], W2[(size_t)d * DD + e], acc);
    dec[(size_t)b * DD + e] = acc;
}

// ---------------------------------------------------------------------------
// Kernel 3: fused  logits[b,n] = vt . tanh(encoded[b,n,:] @ W1 + dec[b,:]) + log(mask)
// 2 waves/block; each wave owns TWO 16-row M tiles staged once in LDS (bf16) and
// reused for all 64 N tiles; 4 B tiles shared across both A tiles -> 8 WMMAs per
// 4 B-fragment loads (2x compute per L2 byte vs 1 M-tile blocking).
// ---------------------------------------------------------------------------
__global__ __launch_bounds__(64) void ptr_attn_main(const float* __restrict__ encoded,
                                                    const __bf16* __restrict__ W1T,
                                                    const float* __restrict__ dec,
                                                    const float* __restrict__ vt,
                                                    const int* __restrict__ mask,
                                                    float* __restrict__ logits) {
    extern __shared__ __bf16 lds[];                 // 2 waves * 32 * ROWSTRIDE bf16
    const int wave = threadIdx.x >> 5;
    const int lane = threadIdx.x & 31;
    __bf16* Atile = lds + (size_t)wave * 32 * ROWSTRIDE;

    const int row0 = blockIdx.x * 64 + wave * 32;   // first of this wave's 32 rows
    const int b    = row0 / NSEQ;                   // 2048 % 64 == 0 -> uniform per block
    const float* decb = dec + (size_t)b * DD;

    // ---- stage A: 32 x 1024 f32 -> bf16 in LDS (vectorized, coalesced) ----
    for (int q = lane; q < 32 * (DD / 4); q += 32) {
        const int r  = q >> 8;                      // 256 quads per row
        const int kq = (q & 255) << 2;
        const float4 v = *(const float4*)(encoded + (size_t)(row0 + r) * DD + kq);
        __bf16* dst = Atile + r * ROWSTRIDE + kq;
        dst[0] = f2bf(v.x); dst[1] = f2bf(v.y); dst[2] = f2bf(v.z); dst[3] = f2bf(v.w);
    }
    __syncthreads();

    const int nl   = lane & 15;
    const int half = lane >> 4;                     // 0 or 1
    const __bf16* arow0 = Atile + nl * ROWSTRIDE + (half << 3);   // A frag, M tile 0
    const __bf16* arow1 = arow0 + 16 * ROWSTRIDE;                 // A frag, M tile 1

    float u0[8], u1[8];
#pragma unroll
    for (int g = 0; g < 8; ++g) { u0[g] = 0.f; u1[g] = 0.f; }

    for (int ng = 0; ng < 16; ++ng) {               // 16 groups x 4 N-tiles = 64 tiles
        const int n0 = ng << 6;
        const __bf16* bbase = W1T + ((size_t)(n0 + nl) << 10) + (half << 3);
        v8f c00 = {}, c01 = {}, c02 = {}, c03 = {};
        v8f c10 = {}, c11 = {}, c12 = {}, c13 = {};
        for (int ks = 0; ks < 32; ++ks) {           // K = 1024 in steps of 32
            const int k0 = ks << 5;
            const v16bf a0 = load_afrag(arow0 + k0);
            const v16bf a1 = load_afrag(arow1 + k0);
            const v16bf b0 = load_afrag(bbase + k0);
            const v16bf b1 = load_afrag(bbase + (16u << 10) + k0);
            const v16bf b2 = load_afrag(bbase + (32u << 10) + k0);
            const v16bf b3 = load_afrag(bbase + (48u << 10) + k0);
            c00 = __builtin_amdgcn_wmma_f32_16x16x32_bf16(false, a0, false, b0, (short)0, c00, false, false);
            c10 = __builtin_amdgcn_wmma_f32_16x16x32_bf16(false, a1, false, b0, (short)0, c10, false, false);
            c01 = __builtin_amdgcn_wmma_f32_16x16x32_bf16(false, a0, false, b1, (short)0, c01, false, false);
            c11 = __builtin_amdgcn_wmma_f32_16x16x32_bf16(false, a1, false, b1, (short)0, c11, false, false);
            c02 = __builtin_amdgcn_wmma_f32_16x16x32_bf16(false, a0, false, b2, (short)0, c02, false, false);
            c12 = __builtin_amdgcn_wmma_f32_16x16x32_bf16(false, a1, false, b2, (short)0, c12, false, false);
            c03 = __builtin_amdgcn_wmma_f32_16x16x32_bf16(false, a0, false, b3, (short)0, c03, false, false);
            c13 = __builtin_amdgcn_wmma_f32_16x16x32_bf16(false, a1, false, b3, (short)0, c13, false, false);
        }
        // epilogue: lane holds col n = n0+t*16+nl; row g+8*half (VGPR g) of each M tile
#define EPILOGUE(T, ACC, U)                                          \
        {                                                            \
            const int n = n0 + ((T) << 4) + nl;                      \
            const float dv = decb[n];                                \
            const float vv = vt[n];                                  \
            _Pragma("unroll")                                        \
            for (int g = 0; g < 8; ++g)                              \
                U[g] += fast_tanh((ACC)[g] + dv) * vv;               \
        }
        EPILOGUE(0, c00, u0) EPILOGUE(1, c01, u0) EPILOGUE(2, c02, u0) EPILOGUE(3, c03, u0)
        EPILOGUE(0, c10, u1) EPILOGUE(1, c11, u1) EPILOGUE(2, c12, u1) EPILOGUE(3, c13, u1)
#undef EPILOGUE
    }

    // reduce each row's partial sums across its 16-lane half
#pragma unroll
    for (int g = 0; g < 8; ++g) {
        float s0 = u0[g], s1 = u1[g];
        s0 += __shfl_xor(s0, 1, 32);  s1 += __shfl_xor(s1, 1, 32);
        s0 += __shfl_xor(s0, 2, 32);  s1 += __shfl_xor(s1, 2, 32);
        s0 += __shfl_xor(s0, 4, 32);  s1 += __shfl_xor(s1, 4, 32);
        s0 += __shfl_xor(s0, 8, 32);  s1 += __shfl_xor(s1, 8, 32);
        u0[g] = s0; u1[g] = s1;
    }
    if (nl == 0) {                                   // lanes 0 (rows +0..7) and 16 (rows +8..15)
        const int rbase = row0 + (half << 3);
#pragma unroll
        for (int g = 0; g < 8; ++g) {
            int r = rbase + g;                       // M tile 0
            int nseq = r & (NSEQ - 1);
            logits[r] = u0[g] + (mask[(size_t)b * NSEQ + nseq] ? 0.0f : LOG_EPS);
            r += 16;                                 // M tile 1
            nseq = r & (NSEQ - 1);
            logits[r] = u1[g] + (mask[(size_t)b * NSEQ + nseq] ? 0.0f : LOG_EPS);
        }
    }
}

// ---------------------------------------------------------------------------
// Kernel 4: per-batch log-softmax over N=2048
// ---------------------------------------------------------------------------
__global__ __launch_bounds__(256) void logsoftmax_kernel(const float* __restrict__ logits,
                                                         float* __restrict__ out) {
    const int b = blockIdx.x;
    const float* row = logits + (size_t)b * NSEQ;
    const int wave = threadIdx.x >> 5, lane = threadIdx.x & 31;
    __shared__ float redm[8], reds[8];

    float m = -INFINITY;
    for (int n = threadIdx.x; n < NSEQ; n += 256) m = fmaxf(m, row[n]);
#pragma unroll
    for (int o = 16; o >= 1; o >>= 1) m = fmaxf(m, __shfl_xor(m, o, 32));
    if (lane == 0) redm[wave] = m;
    __syncthreads();
    float bm = redm[0];
#pragma unroll
    for (int w = 1; w < 8; ++w) bm = fmaxf(bm, redm[w]);

    float s = 0.f;
    for (int n = threadIdx.x; n < NSEQ; n += 256) s += expf(row[n] - bm);
#pragma unroll
    for (int o = 16; o >= 1; o >>= 1) s += __shfl_xor(s, o, 32);
    if (lane == 0) reds[wave] = s;
    __syncthreads();
    float bs = 0.f;
#pragma unroll
    for (int w = 0; w < 8; ++w) bs += reds[w];

    const float lse = bm + logf(bs);
    float* orow = out + (size_t)b * NSEQ;
    for (int n = threadIdx.x; n < NSEQ; n += 256) orow[n] = row[n] - lse;
}

// ---------------------------------------------------------------------------
extern "C" void kernel_launch(void* const* d_in, const int* in_sizes, int n_in,
                              void* d_out, int out_size, void* d_ws, size_t ws_size,
                              hipStream_t stream) {
    const float* encoded = (const float*)d_in[0];   // [32,2048,1024]
    const float* dstate  = (const float*)d_in[1];   // [32,1024]
    const int*   mask    = (const int*)d_in[2];     // [32,2048]
    const float* W1      = (const float*)d_in[3];   // [1024,1024]
    const float* W2      = (const float*)d_in[4];   // [1024,1024]
    const float* vt      = (const float*)d_in[5];   // [1024]
    float* out = (float*)d_out;                     // [32,2048]

    char* ws = (char*)d_ws;
    __bf16* W1T   = (__bf16*)ws;                                   // 2 MB
    float*  dec   = (float*)(ws + (size_t)2 * 1024 * 1024);        // 128 KB
    float*  logit = (float*)(ws + (size_t)2 * 1024 * 1024 + 128 * 1024); // 256 KB

    w1_to_bf16_T<<<dim3(32, 32), dim3(32, 8), 0, stream>>>(W1, W1T);
    dec_gemm<<<dim3(32, 4), 256, 0, stream>>>(dstate, W2, dec);

    const size_t ldsbytes = (size_t)2 * 32 * ROWSTRIDE * sizeof(__bf16); // 132096 B
    (void)hipFuncSetAttribute(reinterpret_cast<const void*>(ptr_attn_main),
                              hipFuncAttributeMaxDynamicSharedMemorySize, (int)ldsbytes);
    ptr_attn_main<<<ROWS / 64, 64, ldsbytes, stream>>>(encoded, W1T, dec, vt, mask, logit);

    logsoftmax_kernel<<<BB, 256, 0, stream>>>(logit, out);
}
